// EncoderBlock_15178414424595
// MI455X (gfx1250) — matrix operand order repaired and checked by reading, not compiled, not run on hardware
//
#include <hip/hip_runtime.h>
#include <math.h>

#define TB   2
#define TT   2048
#define DM   512
#define NH   8
#define DKH  64
#define DFF  2048

typedef __attribute__((ext_vector_type(16))) _Float16 v16h;
typedef __attribute__((ext_vector_type(8)))  float    v8f;

// ---------------------------------------------------------------------------
// WMMA fragment helpers (wave32, v_wmma_f32_16x16x32_f16)
// ---------------------------------------------------------------------------
__device__ __forceinline__ v8f zero8() {
  v8f z = {0.f, 0.f, 0.f, 0.f, 0.f, 0.f, 0.f, 0.f};
  return z;
}

// A fragment: 16x32 f16 tile, row-major, ld in elements.
// lane m=lane&15 holds row m; elems 0..7 -> K = 8*(lane>=16)+0..7,
// elems 8..15 -> K = 16 + 8*(lane>=16)+0..7.
__device__ __forceinline__ v16h load_fragA_h(const _Float16* __restrict__ tile, int ld) {
  const int lane = threadIdx.x & 31;
  const int r    = lane & 15;
  const int kb   = (lane >> 4) << 3;
  const _Float16* p = tile + (size_t)r * ld + kb;
  v16h f;
#pragma unroll
  for (int j = 0; j < 8; ++j) f[j] = p[j];
#pragma unroll
  for (int j = 0; j < 8; ++j) f[8 + j] = p[16 + j];
  return f;
}

// Same as above but source is f32 (attention probabilities), convert on the fly.
__device__ __forceinline__ v16h load_fragA_f32(const float* __restrict__ tile, int ld) {
  const int lane = threadIdx.x & 31;
  const int r    = lane & 15;
  const int kb   = (lane >> 4) << 3;
  const float* p = tile + (size_t)r * ld + kb;
  v16h f;
#pragma unroll
  for (int j = 0; j < 8; ++j) f[j] = (_Float16)p[j];
#pragma unroll
  for (int j = 0; j < 8; ++j) f[8 + j] = (_Float16)p[16 + j];
  return f;
}

// B fragment: 32x16 f16 tile supplied TRANSPOSED (BT is [N, K] row-major).
// lane n=lane&15 holds column n; elems 0..7 -> K = 8*(lane>=16)+0..7,
// elems 8..15 -> K = 16 + 8*(lane>=16)+0..7 (matches A fragment).
__device__ __forceinline__ v16h load_fragB_h(const _Float16* __restrict__ tileT, int ld) {
  const int lane = threadIdx.x & 31;
  const int n    = lane & 15;
  const int kb   = (lane >> 4) << 3;
  const _Float16* p = tileT + (size_t)n * ld + kb;
  v16h f;
#pragma unroll
  for (int e = 0; e < 8; ++e) f[e] = p[e];
#pragma unroll
  for (int e = 0; e < 8; ++e) f[8 + e] = p[16 + e];
  return f;
}

__device__ __forceinline__ v8f wmma16(v16h a, v16h b, v8f c) {
  return __builtin_amdgcn_wmma_f32_16x16x32_f16(
      /*neg_a=*/false, a, /*neg_b=*/false, b,
      /*c_mod=*/(short)0, c, /*reuse_a=*/false, /*reuse_b=*/false);
}

// ---------------------------------------------------------------------------
// Conversion kernels
// ---------------------------------------------------------------------------
__global__ void k_cvt_f16(const float* __restrict__ in, _Float16* __restrict__ out, int n) {
  int i = blockIdx.x * 256 + threadIdx.x;
  if (i < n) out[i] = (_Float16)in[i];
}

// in: [K, N] f32 row-major  ->  out: [N, K] f16 row-major (transposed)
__global__ void k_cvtT_f16(const float* __restrict__ in, _Float16* __restrict__ out,
                           int K, int N) {
  int i = blockIdx.x * 256 + threadIdx.x;
  if (i < K * N) {
    int k = i / N, n = i % N;
    out[(size_t)n * K + k] = (_Float16)in[i];
  }
}

// vh: [B*T, H*DK] (b,t, h*64+d)  ->  vt: [B,H,DK,T]
__global__ void k_transpose_v(const _Float16* __restrict__ vh, _Float16* __restrict__ vt) {
  int i = blockIdx.x * 256 + threadIdx.x;
  if (i >= TB * TT * NH * DKH) return;
  int d = i % DKH;
  int h = (i / DKH) % NH;
  int t = (i / (DKH * NH)) % TT;
  int b = i / (DKH * NH * TT);
  vt[(((size_t)(b * NH + h) * DKH + d) * TT) + t] = vh[i];
}

// ---------------------------------------------------------------------------
// Generic f16 WMMA GEMM: C[M,N] = A[M,K] @ B[K,N] (+bias, +resid, relu)
// A: [M,K] f16 row-major, BT: [N,K] f16 row-major (i.e. B transposed).
// One wave per 16x64 output tile (4 accumulators, A-fragment reused 4x).
// M%16==0, N%64==0, K%32==0 (true for all call sites).
// ---------------------------------------------------------------------------
__global__ void gemm_wmma(const _Float16* __restrict__ A, int lda,
                          const _Float16* __restrict__ BT, int ldb,
                          const float* __restrict__ bias,
                          const float* __restrict__ resid,
                          float* __restrict__ Cf,
                          _Float16* __restrict__ Ch,
                          int M, int N, int K, int relu) {
  const int ncol = N >> 6;
  const int mt = blockIdx.x / ncol;
  const int nt = blockIdx.x % ncol;
  const int m0 = mt << 4, n0 = nt << 6;

  v8f acc[4];
#pragma unroll
  for (int t = 0; t < 4; ++t) acc[t] = zero8();

  const _Float16* Ab = A + (size_t)m0 * lda;
  for (int k0 = 0; k0 < K; k0 += 32) {
    v16h a = load_fragA_h(Ab + k0, lda);
#pragma unroll
    for (int t = 0; t < 4; ++t) {
      v16h b = load_fragB_h(BT + (size_t)(n0 + t * 16) * ldb + k0, ldb);
      acc[t] = wmma16(a, b, acc[t]);
    }
  }

  const int lane = threadIdx.x & 31;
  const int col  = lane & 15;
  const int rb   = (lane >> 4) << 3;
#pragma unroll
  for (int t = 0; t < 4; ++t) {
#pragma unroll
    for (int j = 0; j < 8; ++j) {
      int r = m0 + rb + j;
      int c = n0 + t * 16 + col;
      float v = acc[t][j];
      if (bias)  v += bias[c];
      if (relu)  v = fmaxf(v, 0.0f);
      size_t o = (size_t)r * N + c;
      if (resid) v += resid[o];
      if (Cf) Cf[o] = v;
      if (Ch) Ch[o] = (_Float16)v;
    }
  }
}

// ---------------------------------------------------------------------------
// scores[b,h,q,t] = (q_h . k_h) / sqrt(DK); one wave per 16x16 tile, K=64.
// qh/kh layout: [(b*T+t)*512 + h*64 + d]
// ---------------------------------------------------------------------------
__global__ void scores_wmma(const _Float16* __restrict__ qh,
                            const _Float16* __restrict__ kh,
                            float* __restrict__ attn) {
  const int b = blockIdx.z, h = blockIdx.y;
  const int ntiles = TT >> 4;  // 128
  const int mt = blockIdx.x / ntiles;
  const int nt = blockIdx.x % ntiles;
  const int m0 = mt << 4, n0 = nt << 4;

  const _Float16* Ab = qh + (size_t)(b * TT + m0) * DM + h * DKH;
  const _Float16* Bb = kh + (size_t)(b * TT + n0) * DM + h * DKH;

  v8f acc = zero8();
#pragma unroll
  for (int k0 = 0; k0 < DKH; k0 += 32) {
    v16h a = load_fragA_h(Ab + k0, DM);
    v16h kb = load_fragB_h(Bb + k0, DM);   // cols of K^T = rows of K (contiguous)
    acc = wmma16(a, kb, acc);
  }

  const int lane = threadIdx.x & 31;
  const int col  = lane & 15;
  const int rb   = (lane >> 4) << 3;
  const float scale = 0.125f;  // 1/sqrt(64)
  size_t base = ((size_t)(b * NH + h) * TT + m0) * TT;
#pragma unroll
  for (int j = 0; j < 8; ++j)
    attn[base + (size_t)(rb + j) * TT + n0 + col] = acc[j] * scale;
}

// ---------------------------------------------------------------------------
// Row softmax in place over the last dim (T=2048), one 256-thread block/row.
// ---------------------------------------------------------------------------
__global__ void softmax_rows(float* __restrict__ attn) {
  __shared__ float red[256];
  float* p = attn + (size_t)blockIdx.x * TT;
  const int tid = threadIdx.x;

  float m = -INFINITY;
  for (int i = tid; i < TT; i += 256) m = fmaxf(m, p[i]);
  red[tid] = m; __syncthreads();
  for (int s = 128; s > 0; s >>= 1) {
    if (tid < s) red[tid] = fmaxf(red[tid], red[tid + s]);
    __syncthreads();
  }
  m = red[0]; __syncthreads();

  float sum = 0.f;
  for (int i = tid; i < TT; i += 256) {
    float e = __expf(p[i] - m);
    p[i] = e;
    sum += e;
  }
  red[tid] = sum; __syncthreads();
  for (int s = 128; s > 0; s >>= 1) {
    if (tid < s) red[tid] += red[tid + s];
    __syncthreads();
  }
  float inv = 1.0f / red[0];
  for (int i = tid; i < TT; i += 256) p[i] *= inv;
}

// ---------------------------------------------------------------------------
// ctx[b,h,q,d] = attn[b,h,q,:] @ v[b,h,:,d]; one wave per 16(q)x64(d) tile.
// attn f32 converted to f16 on the fly; vt: [B,H,DK,T] (columns contiguous).
// Output written directly in [B,T, h*64+d] layout (the 0,2,1,3 transpose).
// ---------------------------------------------------------------------------
__global__ void ctx_wmma(const float* __restrict__ attn,
                         const _Float16* __restrict__ vt,
                         _Float16* __restrict__ ctxh) {
  const int b = blockIdx.z, h = blockIdx.y, mt = blockIdx.x;
  const int m0 = mt << 4;

  const float* Ab = attn + ((size_t)(b * NH + h) * TT + m0) * TT;
  const _Float16* Bb = vt + (size_t)(b * NH + h) * DKH * TT;

  v8f acc[4];
#pragma unroll
  for (int t = 0; t < 4; ++t) acc[t] = zero8();

  for (int k0 = 0; k0 < TT; k0 += 32) {
    v16h a = load_fragA_f32(Ab + k0, TT);
#pragma unroll
    for (int t = 0; t < 4; ++t) {
      v16h bf = load_fragB_h(Bb + (size_t)(t * 16) * TT + k0, TT);
      acc[t] = wmma16(a, bf, acc[t]);
    }
  }

  const int lane = threadIdx.x & 31;
  const int col  = lane & 15;
  const int rb   = (lane >> 4) << 3;
#pragma unroll
  for (int t = 0; t < 4; ++t)
#pragma unroll
    for (int j = 0; j < 8; ++j)
      ctxh[(size_t)(b * TT + m0 + rb + j) * DM + h * DKH + t * 16 + col] =
          (_Float16)acc[t][j];
}

// ---------------------------------------------------------------------------
// LayerNorm over D=512, one wave per row, __shfl_xor reductions (wave32).
// ---------------------------------------------------------------------------
__global__ void layernorm_rows(const float* __restrict__ y,
                               const float* __restrict__ g,
                               const float* __restrict__ b,
                               float* __restrict__ outf,
                               _Float16* __restrict__ outh) {
  const int row  = blockIdx.x;
  const int lane = threadIdx.x;
  const float* p = y + (size_t)row * DM;

  float vals[16];
  float s = 0.f;
#pragma unroll
  for (int j = 0; j < 16; ++j) {
    vals[j] = p[lane + 32 * j];
    s += vals[j];
  }
#pragma unroll
  for (int m = 16; m > 0; m >>= 1) s += __shfl_xor(s, m, 32);
  const float mu = s * (1.0f / DM);

  float vs = 0.f;
#pragma unroll
  for (int j = 0; j < 16; ++j) {
    float d = vals[j] - mu;
    vs += d * d;
  }
#pragma unroll
  for (int m = 16; m > 0; m >>= 1) vs += __shfl_xor(vs, m, 32);
  const float inv = rsqrtf(vs * (1.0f / DM) + 1e-5f);

#pragma unroll
  for (int j = 0; j < 16; ++j) {
    int c = lane + 32 * j;
    float o = (vals[j] - mu) * inv * g[c] + b[c];
    outf[(size_t)row * DM + c] = o;
    if (outh) outh[(size_t)row * DM + c] = (_Float16)o;
  }
}

// ---------------------------------------------------------------------------
// Host orchestration
// ---------------------------------------------------------------------------
extern "C" void kernel_launch(void* const* d_in, const int* in_sizes, int n_in,
                              void* d_out, int out_size, void* d_ws, size_t ws_size,
                              hipStream_t stream) {
  (void)in_sizes; (void)n_in; (void)out_size; (void)ws_size;

  const float* x     = (const float*)d_in[0];
  const float* wq_w  = (const float*)d_in[1];
  const float* wq_b  = (const float*)d_in[2];
  const float* wk_w  = (const float*)d_in[3];
  const float* wk_b  = (const float*)d_in[4];
  const float* wv_w  = (const float*)d_in[5];
  const float* wv_b  = (const float*)d_in[6];
  const float* wo_w  = (const float*)d_in[7];
  const float* wo_b  = (const float*)d_in[8];
  const float* ln1_g = (const float*)d_in[9];
  const float* ln1_b = (const float*)d_in[10];
  const float* fc1_w = (const float*)d_in[11];
  const float* fc1_b = (const float*)d_in[12];
  const float* fc2_w = (const float*)d_in[13];
  const float* fc2_b = (const float*)d_in[14];
  const float* ln2_g = (const float*)d_in[15];
  const float* ln2_b = (const float*)d_in[16];

  float* out  = (float*)d_out;                         // [2,2048,512]
  float* attn = out + (size_t)TB * TT * DM;            // [2,8,2048,2048]

  // ---- workspace carve (~74 MB) ----
  char* wsp = (char*)d_ws;
  auto carve = [&](size_t bytes) -> void* {
    void* p = (void*)wsp;
    wsp += (bytes + 255) & ~(size_t)255;
    return p;
  };
  const int Mrows = TB * TT;                           // 4096
  const size_t actH = (size_t)Mrows * DM * sizeof(_Float16);

  _Float16* xh   = (_Float16*)carve(actH);
  _Float16* wqT  = (_Float16*)carve((size_t)DM * DM * 2);
  _Float16* wkT  = (_Float16*)carve((size_t)DM * DM * 2);
  _Float16* wvT  = (_Float16*)carve((size_t)DM * DM * 2);
  _Float16* woT  = (_Float16*)carve((size_t)DM * DM * 2);
  _Float16* fc1T = (_Float16*)carve((size_t)DM * DFF * 2);
  _Float16* fc2T = (_Float16*)carve((size_t)DFF * DM * 2);
  _Float16* qh   = (_Float16*)carve(actH);
  _Float16* kh   = (_Float16*)carve(actH);
  _Float16* vh   = (_Float16*)carve(actH);
  _Float16* vt   = (_Float16*)carve(actH);
  _Float16* ctxh = (_Float16*)carve(actH);
  _Float16* hh   = (_Float16*)carve(actH);
  _Float16* uh   = (_Float16*)carve((size_t)Mrows * DFF * 2);
  float*    y1   = (float*)carve((size_t)Mrows * DM * 4);
  float*    hbuf = (float*)carve((size_t)Mrows * DM * 4);
  float*    y2   = (float*)carve((size_t)Mrows * DM * 4);

  const dim3 w32(32);
  const int nX = Mrows * DM;  // 2097152

  // 1) precision converts (weights pre-transposed for contiguous B fragments)
  k_cvt_f16 <<<nX / 256, 256, 0, stream>>>(x, xh, nX);
  k_cvtT_f16<<<(DM * DM)  / 256, 256, 0, stream>>>(wq_w,  wqT,  DM,  DM);
  k_cvtT_f16<<<(DM * DM)  / 256, 256, 0, stream>>>(wk_w,  wkT,  DM,  DM);
  k_cvtT_f16<<<(DM * DM)  / 256, 256, 0, stream>>>(wv_w,  wvT,  DM,  DM);
  k_cvtT_f16<<<(DM * DM)  / 256, 256, 0, stream>>>(wo_w,  woT,  DM,  DM);
  k_cvtT_f16<<<(DM * DFF) / 256, 256, 0, stream>>>(fc1_w, fc1T, DM,  DFF);
  k_cvtT_f16<<<(DFF * DM) / 256, 256, 0, stream>>>(fc2_w, fc2T, DFF, DM);

  // 2) Q/K/V projections (f16 out), V transposed to [B,H,DK,T]
  const int gProj = (Mrows / 16) * (DM / 64);
  gemm_wmma<<<gProj, w32, 0, stream>>>(xh, DM, wqT, DM, wq_b, nullptr, nullptr, qh,
                                       Mrows, DM, DM, 0);
  gemm_wmma<<<gProj, w32, 0, stream>>>(xh, DM, wkT, DM, wk_b, nullptr, nullptr, kh,
                                       Mrows, DM, DM, 0);
  gemm_wmma<<<gProj, w32, 0, stream>>>(xh, DM, wvT, DM, wv_b, nullptr, nullptr, vh,
                                       Mrows, DM, DM, 0);
  k_transpose_v<<<nX / 256, 256, 0, stream>>>(vh, vt);

  // 3) scores -> d_out attn region, 4) softmax in place
  scores_wmma<<<dim3((TT / 16) * (TT / 16), NH, TB), w32, 0, stream>>>(qh, kh, attn);
  softmax_rows<<<TB * NH * TT, 256, 0, stream>>>(attn);

  // 5) ctx = attn @ V (f32->f16 on the fly), stored in [B,T,H*DK] layout
  ctx_wmma<<<dim3(TT / 16, NH, TB), w32, 0, stream>>>(attn, vt, ctxh);

  // 6) attn_out = ctx @ Wo + b + x; LN1 -> h (f32 + f16)
  gemm_wmma<<<gProj, w32, 0, stream>>>(ctxh, DM, woT, DM, wo_b, x, y1, nullptr,
                                       Mrows, DM, DM, 0);
  layernorm_rows<<<Mrows, w32, 0, stream>>>(y1, ln1_g, ln1_b, hbuf, hh);

  // 7) FFN: u = relu(h@fc1+b1); y2 = u@fc2+b2 + h; LN2 -> out
  gemm_wmma<<<(Mrows / 16) * (DFF / 64), w32, 0, stream>>>(
      hh, DM, fc1T, DM, fc1_b, nullptr, nullptr, uh, Mrows, DFF, DM, 1);
  gemm_wmma<<<gProj, w32, 0, stream>>>(uh, DFF, fc2T, DFF, fc2_b, hbuf, y2, nullptr,
                                       Mrows, DM, DFF, 0);
  layernorm_rows<<<Mrows, w32, 0, stream>>>(y2, ln2_g, ln2_b, out, nullptr);
}
